// Lattice_Transformer_SeqLabel_24730421690488
// MI455X (gfx1250) — matrix operand order repaired and checked by reading, not compiled, not run on hardware
//
#include <hip/hip_runtime.h>

// ---------------------------------------------------------------------------
// Lattice Transformer forward for MI455X (gfx1250, wave32, WMMA).
//
// BD[b,h,q,k] = sum_c rel[b,q,k,c]*m[b,h,q,c], m = (q+vb) @ Wr_headslice^T,
// contracted query-side first: 32x fewer FLOPs than the reference rel@Wr and
// never materializes the 268MB rel tensor (rows regenerated from the 4.2MB
// L2-resident T tables straight into WMMA A-fragments).
// All GEMMs: bf16 inputs, v_wmma_f32_16x16x32_bf16, fp32 accumulate.
// B operands stored transposed [N][K]: each lane's fragment slice is two runs of 8
// contiguous bf16, 16 apart in K -> two aligned 16B loads, no predication.
// Waves own 32x32 macro-tiles (2x2 register blocking): 4 WMMAs per pair of
// A/B fragment loads -> 2x the arithmetic intensity of 1-tile-per-wave.
// ---------------------------------------------------------------------------

#define B_    4
#define S_    200
#define L_    256
#define D_    256
#define H_    8
#define DH_   32
#define FF_   1024
#define LBL_  20
#define NROW  1024          // B*L
#define PE_N  1025          // 2*MAXP+1
#define NEGV  (-1e15f)

typedef __attribute__((ext_vector_type(16))) __bf16 bf16x16;
typedef __attribute__((ext_vector_type(8)))  __bf16 bf16x8;
typedef __attribute__((ext_vector_type(8)))  float  f32x8;

// ---------------- wave helpers (wave32) ----------------
__device__ inline float wave_sum(float v) {
  #pragma unroll
  for (int m = 16; m; m >>= 1) v += __shfl_xor(v, m, 32);
  return v;
}
__device__ inline float wave_max(float v) {
  #pragma unroll
  for (int m = 16; m; m >>= 1) v = fmaxf(v, __shfl_xor(v, m, 32));
  return v;
}

// ---------------- WMMA fragment loads (documented gfx1250 layouts) ---------
// A 16x32 bf16, row-major A[lda]: lane<16 -> M=lane, K halves {0..7,16..23};
// lane>=16 -> M=lane-16, K halves {8..15,24..31}.  Chunks are 16B-aligned.
__device__ inline bf16x16 load_a_frag(const __bf16* __restrict__ A, int lda,
                                      int row0, int k0, int lane) {
  const __bf16* p = A + (long)(row0 + (lane & 15)) * lda + k0 + ((lane < 16) ? 0 : 8);
  bf16x8 lo = *(const bf16x8*)p;
  bf16x8 hi = *(const bf16x8*)(p + 16);
  bf16x16 r;
  #pragma unroll
  for (int i = 0; i < 8; i++) { r[i] = lo[i]; r[8 + i] = hi[i]; }
  return r;
}
// B 32x16 bf16 from TRANSPOSED storage Bt[N][K]: element (k,n) = Bt[n*ldb+k].
// lane<16 -> N=lane, K halves {0..7,16..23}; lane>=16 -> K halves {8..15,24..31}.
__device__ inline bf16x16 load_bt_frag(const __bf16* __restrict__ Bt, int ldb,
                                       int k0, int n0, int lane) {
  const __bf16* p = Bt + (long)(n0 + (lane & 15)) * ldb + k0 + ((lane < 16) ? 0 : 8);
  bf16x8 lo = *(const bf16x8*)p;
  bf16x8 hi = *(const bf16x8*)(p + 16);
  bf16x16 r;
  #pragma unroll
  for (int i = 0; i < 8; i++) { r[i] = lo[i]; r[8 + i] = hi[i]; }
  return r;
}

// ---------------- generic batched WMMA GEMM (B transposed, 2x2 blocked) ----
// C[m,n] = sum_k A[m,k]*Bt[n,k] (+bias[n]) (+relu).  One 32x32 macro-tile per
// wave (4 accumulators), 8 waves per block.  blockIdx.y/z batch via strides.
// Requires M % 32 == 0; Bt padded so all n in [0, 32*ceil(N/32)) load cleanly.
template <bool RELU>
__global__ void gemm_wmma_kernel(const __bf16* __restrict__ A,
                                 const __bf16* __restrict__ Bt,
                                 const float* __restrict__ bias,
                                 float* __restrict__ C, __bf16* __restrict__ Cbf,
                                 int M, int N, int K, int lda, int ldb, int ldc,
                                 long sAy, long sAz, long sBy, long sBz,
                                 long sCy, long sCz) {
  int lane = threadIdx.x & 31, wid = threadIdx.x >> 5;
  int MT2 = M >> 5;                               // 32-row macro tiles
  int NT2 = (N + 31) >> 5;                        // 32-col macro tiles
  int tile = blockIdx.x * 8 + wid;
  if (tile >= MT2 * NT2) return;                  // wave-uniform
  A  += blockIdx.y * sAy + blockIdx.z * sAz;
  Bt += blockIdx.y * sBy + blockIdx.z * sBz;
  long coff = blockIdx.y * sCy + blockIdx.z * sCz;
  int m0 = (tile / NT2) * 32, n0 = (tile % NT2) * 32;
  f32x8 acc00 = {}, acc01 = {}, acc10 = {}, acc11 = {};
  for (int k0 = 0; k0 < K; k0 += 32) {
    bf16x16 a0 = load_a_frag(A, lda, m0,      k0, lane);
    bf16x16 a1 = load_a_frag(A, lda, m0 + 16, k0, lane);
    bf16x16 b0 = load_bt_frag(Bt, ldb, k0, n0,      lane);
    bf16x16 b1 = load_bt_frag(Bt, ldb, k0, n0 + 16, lane);
    acc00 = __builtin_amdgcn_wmma_f32_16x16x32_bf16(false, a0, false, b0, (short)0, acc00, false, false);
    acc01 = __builtin_amdgcn_wmma_f32_16x16x32_bf16(false, a0, false, b1, (short)0, acc01, false, false);
    acc10 = __builtin_amdgcn_wmma_f32_16x16x32_bf16(false, a1, false, b0, (short)0, acc10, false, false);
    acc11 = __builtin_amdgcn_wmma_f32_16x16x32_bf16(false, a1, false, b1, (short)0, acc11, false, false);
  }
  #pragma unroll
  for (int tn = 0; tn < 2; tn++) {
    int col = n0 + tn * 16 + (lane & 15);
    if (col >= N) continue;
    float bi = bias ? bias[col] : 0.f;
    #pragma unroll
    for (int tm = 0; tm < 2; tm++) {
      const f32x8& acc = tm ? (tn ? acc11 : acc10) : (tn ? acc01 : acc00);
      #pragma unroll
      for (int j = 0; j < 8; j++) {
        int row = m0 + tm * 16 + ((lane < 16) ? j : j + 8);
        float v = acc[j] + bi;
        if (RELU) v = fmaxf(v, 0.f);
        long idx = coff + (long)row * ldc + col;
        if (C)   C[idx]   = v;
        if (Cbf) Cbf[idx] = (__bf16)v;
      }
    }
  }
}

// ---------------- fused BD kernel ------------------------------------------
// Regenerates rel rows (4 gathers from Tproj + bias + relu) straight into
// WMMA A-fragments and contracts against m (bf16, padded to 16 "heads" so
// B-fragment loads are unconditional).  Each wave handles TWO k-tiles sharing
// one B-fragment per step (2 WMMAs / B-load).  Merges with AC scores
// in-place: score = (AC + BD + bdb)*scale, masked to NEG past tot.
__global__ void bd_fused_kernel(const float* __restrict__ Tproj,   // [4][1025][256]
                                const float* __restrict__ b_fus,   // [256]
                                const int* __restrict__ pos_s, const int* __restrict__ pos_e,
                                const int* __restrict__ seq_len, const int* __restrict__ lex_num,
                                const __bf16* __restrict__ m_bf,   // [16][1024][256], rows 8..15 zero
                                const float* __restrict__ bdb,     // [1024][8]
                                float* __restrict__ scores) {      // [B*H][256][256] in/out
  int lane = threadIdx.x & 31, wid = threadIdx.x >> 5;
  int t = blockIdx.x * 8 + wid;          // 0..8191
  int bq = t >> 3, kp = t & 7;           // kp: pair of k-tiles (rows kp*32..+31)
  int b = bq >> 8, q = bq & 255;
  int ps_q = pos_s[b * L_ + q], pe_q = pos_e[b * L_ + q];
  // gather pointers for this lane's two A rows (k and k+16)
  const float* tp[2][4];
  #pragma unroll
  for (int r = 0; r < 2; r++) {
    int krow = kp * 32 + r * 16 + (lane & 15);
    int ps_k = pos_s[b * L_ + krow], pe_k = pos_e[b * L_ + krow];
    tp[r][0] = Tproj + (long)(ps_q - ps_k + 512) * 256;
    tp[r][1] = Tproj + (long)PE_N * 256     + (long)(ps_q - pe_k + 512) * 256;
    tp[r][2] = Tproj + (long)2 * PE_N * 256 + (long)(pe_q - ps_k + 512) * 256;
    tp[r][3] = Tproj + (long)3 * PE_N * 256 + (long)(pe_q - pe_k + 512) * 256;
  }
  int kb = (lane < 16) ? 0 : 8;
  int h = lane & 15;                     // B-matrix column (head, 8..15 padded)
  const __bf16* mrow = m_bf + (long)h * NROW * 256 + (long)bq * 256;
  f32x8 acc0 = {}, acc1 = {};
  for (int c0 = 0; c0 < 256; c0 += 32) {
    // one shared B-fragment (m row slice), two A-fragments
    const __bf16* p = mrow + c0 + ((lane < 16) ? 0 : 8);
    bf16x8 lo = *(const bf16x8*)p;
    bf16x8 hi = *(const bf16x8*)(p + 16);
    bf16x16 bm;
    #pragma unroll
    for (int i = 0; i < 8; i++) { bm[i] = lo[i]; bm[8 + i] = hi[i]; }
    bf16x16 a0, a1;
    #pragma unroll
    for (int i = 0; i < 8; i++) {
      int c = c0 + kb + i, c2 = c0 + 16 + kb + i;
      float bf1 = b_fus[c], bf2 = b_fus[c2];
      float v0  = tp[0][0][c]  + tp[0][1][c]  + tp[0][2][c]  + tp[0][3][c]  + bf1;
      float v02 = tp[0][0][c2] + tp[0][1][c2] + tp[0][2][c2] + tp[0][3][c2] + bf2;
      float v1  = tp[1][0][c]  + tp[1][1][c]  + tp[1][2][c]  + tp[1][3][c]  + bf1;
      float v12 = tp[1][0][c2] + tp[1][1][c2] + tp[1][2][c2] + tp[1][3][c2] + bf2;
      a0[i] = (__bf16)fmaxf(v0, 0.f);  a0[8 + i] = (__bf16)fmaxf(v02, 0.f);
      a1[i] = (__bf16)fmaxf(v1, 0.f);  a1[8 + i] = (__bf16)fmaxf(v12, 0.f);
    }
    acc0 = __builtin_amdgcn_wmma_f32_16x16x32_bf16(false, a0, false, bm, (short)0, acc0, false, false);
    acc1 = __builtin_amdgcn_wmma_f32_16x16x32_bf16(false, a1, false, bm, (short)0, acc1, false, false);
  }
  if (h >= H_) return;                   // lanes for padded heads carry no output
  int tot = seq_len[b] + lex_num[b];
  float bdbv = bdb[bq * H_ + h];
  const float scale = 0.1767766952966369f;   // 1/sqrt(32)
  float* srow = scores + ((long)(b * H_ + h) * 256 + q) * 256;
  #pragma unroll
  for (int r = 0; r < 2; r++) {
    const f32x8& acc = r ? acc1 : acc0;
    #pragma unroll
    for (int j = 0; j < 8; j++) {
      int k = kp * 32 + r * 16 + ((lane < 16) ? j : j + 8);
      float s = (srow[k] + acc[j] + bdbv) * scale;
      srow[k] = (k < tot) ? s : NEGV;
    }
  }
}

// ---------------- small VALU kernels ---------------------------------------
__global__ void cvt_bf16_kernel(const float* __restrict__ src,
                                __bf16* __restrict__ dst, int n) {
  int i = blockIdx.x * 256 + threadIdx.x;
  if (i < n) dst[i] = (__bf16)src[i];
}
// transpose-convert: src f32 [K][N] -> dst bf16 [Npad][K], zero-padded rows.
__global__ void cvt_t_bf16_kernel(const float* __restrict__ src,
                                  __bf16* __restrict__ dst, int K, int N, int Npad) {
  int i = blockIdx.x * 256 + threadIdx.x;
  if (i >= Npad * K) return;
  int n = i / K, k = i % K;
  dst[i] = (n < N) ? (__bf16)src[(long)k * N + n] : (__bf16)0.f;
}
__global__ void zero_bf16_kernel(__bf16* __restrict__ p, int n) {
  int i = blockIdx.x * 256 + threadIdx.x;
  if (i < n) p[i] = (__bf16)0.f;
}
// V f32 [1024][256] -> Vt bf16 [(b*8+h)*32+d][256] = V[(b*256+kpos)][h*32+d]
__global__ void vtrans_kernel(const float* __restrict__ V, __bf16* __restrict__ Vt) {
  int i = blockIdx.x * 256 + threadIdx.x;     // over 1024*256
  int kpos = i & 255, r = i >> 8;
  int d = r & 31, h = (r >> 5) & 7, b = r >> 8;
  Vt[(long)r * 256 + kpos] = (__bf16)V[((long)b * 256 + kpos) * 256 + h * 32 + d];
}

// Tproj[kk][p][t] = sum_e pe[p][e] * W_fus[kk*256+e][t]
__global__ void peproj_kernel(const float* __restrict__ W_fus,
                              float* __restrict__ Tproj) {
  int p = blockIdx.x, kk = blockIdx.y, t = threadIdx.x;
  __shared__ float pe[256];
  float fr = expf((float)(t & 127) * (-9.210340371976184f / 127.f)); // ln(1e4)
  pe[t] = (t < 128) ? sinf((float)p * fr) : cosf((float)p * fr);
  __syncthreads();
  float s = 0.f;
  for (int e = 0; e < 256; e++) s += pe[e] * W_fus[(long)(kk * 256 + e) * 256 + t];
  Tproj[((long)kk * PE_N + p) * 256 + t] = s;
}

__global__ void embed_kernel(const int* __restrict__ lattice, const int* __restrict__ bigrams,
                             const int* __restrict__ seq_len, const int* __restrict__ lex_num,
                             const float* __restrict__ bert, const float* __restrict__ lat_t,
                             const float* __restrict__ big_t,
                             const float* __restrict__ Wc, const float* __restrict__ bc,
                             const float* __restrict__ Wl, const float* __restrict__ bl,
                             float* __restrict__ x, __bf16* __restrict__ xbf) {
  int i = blockIdx.x, b = i >> 8, l = i & 255, t = threadIdx.x;
  __shared__ float row[868];
  int lat = lattice[i];
  bool inS = (l < S_);
  int bg = inS ? bigrams[b * S_ + l] : 0;
  for (int j = t; j < 50; j += 256) {
    row[j]      = lat_t[(long)lat * 50 + j];
    row[50 + j] = inS ? big_t[(long)bg * 50 + j] : 0.f;
  }
  for (int j = t; j < 768; j += 256)
    row[100 + j] = inS ? bert[((long)b * S_ + l) * 768 + j] : 0.f;
  __syncthreads();
  int sl = seq_len[b], ln = lex_num[b];
  bool cm = (l < sl), lm = (l < sl + ln) && !cm;
  float v = 0.f;
  if (cm) { v = bc[t]; for (int e = 0; e < 868; e++) v += row[e] * Wc[(long)e * 256 + t]; }
  else if (lm) { v = bl[t]; for (int e = 0; e < 50; e++) v += row[e] * Wl[(long)e * 256 + t]; }
  x[(long)i * 256 + t] = v;
  xbf[(long)i * 256 + t] = (__bf16)v;
}

// Qu = bf(Q + u), Qv = bf(Q + vb), bdb[i][h] = sum_d Qv[i][h*32+d]*br[h*32+d]
__global__ void quqv_kernel(const float* __restrict__ Q, const float* __restrict__ u,
                            const float* __restrict__ vb, const float* __restrict__ br,
                            __bf16* __restrict__ Qu, __bf16* __restrict__ Qv,
                            float* __restrict__ bdb) {
  int i = blockIdx.x, c = threadIdx.x;
  float qf = Q[(long)i * 256 + c];
  float qv = qf + vb[c];
  Qu[(long)i * 256 + c] = (__bf16)(qf + u[c]);
  Qv[(long)i * 256 + c] = (__bf16)qv;
  float p = wave_sum(qv * br[c]);          // wave w == head h (c = 32h + d)
  if ((c & 31) == 0) bdb[i * H_ + (c >> 5)] = p;
}

// wave-per-row softmax over 256 -> bf16 probabilities
__global__ void softmax_kernel(const float* __restrict__ scores,
                               __bf16* __restrict__ P) {
  int lane = threadIdx.x & 31, wid = threadIdx.x >> 5;
  long row = (long)blockIdx.x * 8 + wid;
  const float* s = scores + row * 256;
  float v[8], mx = -3.4e38f;
  #pragma unroll
  for (int i = 0; i < 8; i++) { v[i] = s[lane + i * 32]; mx = fmaxf(mx, v[i]); }
  mx = wave_max(mx);
  float sum = 0.f;
  #pragma unroll
  for (int i = 0; i < 8; i++) { v[i] = __expf(v[i] - mx); sum += v[i]; }
  sum = wave_sum(sum);
  float inv = 1.f / sum;
  #pragma unroll
  for (int i = 0; i < 8; i++) P[row * 256 + lane + i * 32] = (__bf16)(v[i] * inv);
}

// x = LN(a + x)*g + b  (wave per row), writes f32 + bf16 views
__global__ void add_ln_kernel(const float* __restrict__ a, float* __restrict__ x,
                              __bf16* __restrict__ xbf, const float* __restrict__ g,
                              const float* __restrict__ bt) {
  int lane = threadIdx.x & 31, wid = threadIdx.x >> 5;
  long row = (long)blockIdx.x * 8 + wid;
  float v[8], s = 0.f;
  #pragma unroll
  for (int i = 0; i < 8; i++) { v[i] = a[row * 256 + lane + i * 32] + x[row * 256 + lane + i * 32]; s += v[i]; }
  float mean = wave_sum(s) * (1.f / 256.f);
  float vs = 0.f;
  #pragma unroll
  for (int i = 0; i < 8; i++) { float d = v[i] - mean; vs += d * d; }
  float var = wave_sum(vs) * (1.f / 256.f);
  float r = rsqrtf(var + 1e-5f);
  #pragma unroll
  for (int i = 0; i < 8; i++) {
    int c = lane + i * 32;
    float y = (v[i] - mean) * r * g[c] + bt[c];
    x[row * 256 + c] = y;
    xbf[row * 256 + c] = (__bf16)y;
  }
}

__global__ void gather_out_kernel(const float* __restrict__ xout, float* __restrict__ out) {
  int i = blockIdx.x * 256 + threadIdx.x;
  if (i >= B_ * S_ * LBL_) return;
  int c = i % LBL_, s = (i / LBL_) % S_, b = i / (S_ * LBL_);
  out[i] = xout[((long)(b * L_ + s)) * LBL_ + c];
}

// ---------------------------------------------------------------------------
extern "C" void kernel_launch(void* const* d_in, const int* in_sizes, int n_in,
                              void* d_out, int out_size, void* d_ws, size_t ws_size,
                              hipStream_t stream) {
  (void)in_sizes; (void)n_in; (void)out_size; (void)ws_size;
  const int*   lattice = (const int*)  d_in[0];
  const int*   bigrams = (const int*)  d_in[1];
  const int*   seq_len = (const int*)  d_in[2];
  const int*   lex_num = (const int*)  d_in[3];
  const int*   pos_s   = (const int*)  d_in[4];
  const int*   pos_e   = (const int*)  d_in[5];
  const float* bert    = (const float*)d_in[7];
  const float* lat_t   = (const float*)d_in[8];
  const float* big_t   = (const float*)d_in[9];
  const float* Wc  = (const float*)d_in[10]; const float* bc  = (const float*)d_in[11];
  const float* Wl  = (const float*)d_in[12]; const float* bl  = (const float*)d_in[13];
  const float* Wfus= (const float*)d_in[14]; const float* bfus= (const float*)d_in[15];
  const float* Wq  = (const float*)d_in[16]; const float* bq  = (const float*)d_in[17];
  const float* Wk  = (const float*)d_in[18]; const float* bk  = (const float*)d_in[19];
  const float* Wv  = (const float*)d_in[20]; const float* bv  = (const float*)d_in[21];
  const float* Wr  = (const float*)d_in[22]; const float* br  = (const float*)d_in[23];
  const float* u_  = (const float*)d_in[24]; const float* vb_ = (const float*)d_in[25];
  const float* Wo  = (const float*)d_in[26]; const float* bo  = (const float*)d_in[27];
  const float* l1g = (const float*)d_in[28]; const float* l1b = (const float*)d_in[29];
  const float* W1  = (const float*)d_in[30]; const float* b1  = (const float*)d_in[31];
  const float* W2  = (const float*)d_in[32]; const float* b2  = (const float*)d_in[33];
  const float* l2g = (const float*)d_in[34]; const float* l2b = (const float*)d_in[35];
  const float* Wout= (const float*)d_in[36]; const float* bout= (const float*)d_in[37];
  float* out = (float*)d_out;

  // ---- workspace carve (256B aligned) ----
  char* w = (char*)d_ws;
  auto carve = [&](size_t bytes) { void* p = w; w += (bytes + 255) & ~size_t(255); return p; };
  float*  Tproj = (float*) carve(sizeof(float) * 4 * PE_N * 256);
  float*  x     = (float*) carve(sizeof(float) * NROW * 256);
  __bf16* xbf   = (__bf16*)carve(sizeof(__bf16) * NROW * 256);
  float*  Q     = (float*) carve(sizeof(float) * NROW * 256);
  __bf16* Qu    = (__bf16*)carve(sizeof(__bf16) * NROW * 256);
  __bf16* Qv    = (__bf16*)carve(sizeof(__bf16) * NROW * 256);
  __bf16* Kbf   = (__bf16*)carve(sizeof(__bf16) * NROW * 256);
  float*  Vf    = (float*) carve(sizeof(float) * NROW * 256);
  __bf16* Vt    = (__bf16*)carve(sizeof(__bf16) * NROW * 256);
  float*  bdb   = (float*) carve(sizeof(float) * NROW * H_);
  __bf16* m_bf  = (__bf16*)carve(sizeof(__bf16) * 16 * NROW * 256);  // padded heads
  float*  scores= (float*) carve(sizeof(float) * B_ * H_ * 256 * 256);
  __bf16* P     = (__bf16*)carve(sizeof(__bf16) * B_ * H_ * 256 * 256);
  __bf16* Obf   = (__bf16*)carve(sizeof(__bf16) * NROW * 256);
  float*  tmp1  = (float*) carve(sizeof(float) * NROW * 256);
  __bf16* h1bf  = (__bf16*)carve(sizeof(__bf16) * NROW * FF_);
  float*  xout  = (float*) carve(sizeof(float) * NROW * LBL_);
  __bf16* Wqb   = (__bf16*)carve(sizeof(__bf16) * 2 * 256 * 256);   // transposed
  __bf16* Wkb   = (__bf16*)carve(sizeof(__bf16) * 2 * 256 * 256);   // transposed
  __bf16* Wvb   = (__bf16*)carve(sizeof(__bf16) * 2 * 256 * 256);   // transposed
  __bf16* Wrb   = (__bf16*)carve(sizeof(__bf16) * 2 * 256 * 256);   // plain (BT access)
  __bf16* Wob   = (__bf16*)carve(sizeof(__bf16) * 2 * 256 * 256);   // transposed
  __bf16* W1b   = (__bf16*)carve(sizeof(__bf16) * 2 * FF_ * 256);   // transposed [1024][256]
  __bf16* W2b   = (__bf16*)carve(sizeof(__bf16) * 2 * 256 * FF_);   // transposed [256][1024]
  __bf16* Woutb = (__bf16*)carve(sizeof(__bf16) * 32 * 256);        // transposed, padded N=32

  auto cvtT = [&](const float* s, __bf16* d, int K, int N, int Npad) {
    cvt_t_bf16_kernel<<<(Npad * K + 255) / 256, 256, 0, stream>>>(s, d, K, N, Npad);
  };
  for (int l = 0; l < 2; l++) {
    cvtT(Wq + l * 65536, Wqb + l * 65536, 256, 256, 256);
    cvtT(Wk + l * 65536, Wkb + l * 65536, 256, 256, 256);
    cvtT(Wv + l * 65536, Wvb + l * 65536, 256, 256, 256);
    cvtT(Wo + l * 65536, Wob + l * 65536, 256, 256, 256);
    cvtT(W1 + (long)l * 256 * FF_, W1b + (long)l * FF_ * 256, 256, FF_, FF_);
    cvtT(W2 + (long)l * FF_ * 256, W2b + (long)l * 256 * FF_, FF_, 256, 256);
  }
  cvt_bf16_kernel<<<(2 * 65536 + 255) / 256, 256, 0, stream>>>(Wr, Wrb, 2 * 65536);
  cvtT(Wout, Woutb, 256, LBL_, 32);
  // zero padded head rows 8..15 of m (once; m GEMM only writes rows 0..7)
  zero_bf16_kernel<<<(8 * NROW * 256) / 256, 256, 0, stream>>>(
      m_bf + (long)8 * NROW * 256, 8 * NROW * 256);

  peproj_kernel<<<dim3(PE_N, 4, 1), 256, 0, stream>>>(Wfus, Tproj);
  embed_kernel<<<NROW, 256, 0, stream>>>(lattice, bigrams, seq_len, lex_num, bert,
                                         lat_t, big_t, Wc, bc, Wl, bl, x, xbf);

  auto gx = [](int M, int N) { return ((M / 32) * ((N + 31) / 32) + 7) / 8; };

  for (int l = 0; l < 2; l++) {
    const long WO = (long)l * 65536;
    // Q/K/V projections: [1024,256] @ [256,256]  (Bt = transposed weights)
    gemm_wmma_kernel<false><<<gx(NROW, 256), 256, 0, stream>>>(
        xbf, Wqb + WO, bq + l * 256, Q, nullptr, NROW, 256, 256, 256, 256, 256,
        0, 0, 0, 0, 0, 0);
    gemm_wmma_kernel<false><<<gx(NROW, 256), 256, 0, stream>>>(
        xbf, Wkb + WO, bk + l * 256, nullptr, Kbf, NROW, 256, 256, 256, 256, 256,
        0, 0, 0, 0, 0, 0);
    gemm_wmma_kernel<false><<<gx(NROW, 256), 256, 0, stream>>>(
        xbf, Wvb + WO, bv + l * 256, Vf, nullptr, NROW, 256, 256, 256, 256, 256,
        0, 0, 0, 0, 0, 0);
    vtrans_kernel<<<NROW, 256, 0, stream>>>(Vf, Vt);
    quqv_kernel<<<NROW, 256, 0, stream>>>(Q, u_ + l * 256, vb_ + l * 256,
                                          br + l * 256, Qu, Qv, bdb);
    // m[h] = Qv[:,h*32:+32] @ Wr[:, h*32:+32]^T  (Wr plain = BT access; K=32)
    gemm_wmma_kernel<false><<<dim3(gx(NROW, 256), H_, 1), 256, 0, stream>>>(
        Qv, Wrb + WO, nullptr, nullptr, m_bf, NROW, 256, 32, 256, 256, 256,
        /*sAy*/32, 0, /*sBy*/32, 0, /*sCy*/(long)NROW * 256, 0);
    // AC = Qu_h @ K_h^T  (K plain row-major = BT access; K=32, one WMMA step)
    gemm_wmma_kernel<false><<<dim3(gx(256, 256), H_, B_), 256, 0, stream>>>(
        Qu, Kbf, nullptr, scores, nullptr, 256, 256, 32, 256, 256, 256,
        /*sAy*/32, /*sAz*/(long)L_ * 256, /*sBy*/32, /*sBz*/(long)L_ * 256,
        /*sCy*/(long)256 * 256, /*sCz*/(long)H_ * 256 * 256);
    // BD fused: regenerate rel rows, WMMA vs m, merge scores + mask + scale
    bd_fused_kernel<<<1024, 256, 0, stream>>>(Tproj, bfus, pos_s, pos_e, seq_len,
                                              lex_num, m_bf, bdb, scores);
    softmax_kernel<<<(B_ * H_ * 256) / 8, 256, 0, stream>>>(scores, P);
    // O = P @ V_h   (Vt = transposed V per (b,h))
    gemm_wmma_kernel<false><<<dim3(gx(256, 32), H_, B_), 256, 0, stream>>>(
        P, Vt, nullptr, nullptr, Obf, 256, 32, 256, 256, 256, 256,
        /*sAy*/(long)256 * 256, /*sAz*/(long)H_ * 256 * 256,
        /*sBy*/(long)32 * 256, /*sBz*/(long)H_ * 32 * 256,
        /*sCy*/32, /*sCz*/(long)L_ * 256);
    // Wo projection + residual LN1
    gemm_wmma_kernel<false><<<gx(NROW, 256), 256, 0, stream>>>(
        Obf, Wob + WO, bo + l * 256, tmp1, nullptr, NROW, 256, 256, 256, 256, 256,
        0, 0, 0, 0, 0, 0);
    add_ln_kernel<<<NROW / 8, 256, 0, stream>>>(tmp1, x, xbf, l1g + l * 256, l1b + l * 256);
    // FFN
    gemm_wmma_kernel<true><<<gx(NROW, FF_), 256, 0, stream>>>(
        xbf, W1b + (long)l * FF_ * 256, b1 + l * FF_, nullptr, h1bf,
        NROW, FF_, 256, 256, 256, FF_, 0, 0, 0, 0, 0, 0);
    gemm_wmma_kernel<false><<<gx(NROW, 256), 256, 0, stream>>>(
        h1bf, W2b + (long)l * 256 * FF_, b2 + l * 256, tmp1, nullptr,
        NROW, 256, FF_, FF_, FF_, 256, 0, 0, 0, 0, 0, 0);
    add_ln_kernel<<<NROW / 8, 256, 0, stream>>>(tmp1, x, xbf, l2g + l * 256, l2b + l * 256);
  }
  // output projection (Bt padded to 32 rows) + gather rows s<S
  gemm_wmma_kernel<false><<<gx(NROW, LBL_), 256, 0, stream>>>(
      xbf, Woutb, bout, xout, nullptr, NROW, LBL_, 256, 256, 256, LBL_,
      0, 0, 0, 0, 0, 0);
  gather_out_kernel<<<(B_ * S_ * LBL_ + 255) / 256, 256, 0, stream>>>(xout, out);
}
